// attention_layer_45054206935606
// MI455X (gfx1250) — matrix operand order repaired and checked by reading, not compiled, not run on hardware
//
#include <hip/hip_runtime.h>

// ---------------------------------------------------------------------------
// Transformer block (pre-LN attention + MLP) for MI455X / gfx1250.
// Memory-bound (~0.4 TFLOP vs ~2.2 GB HBM -> ~95us floor at 23.3 TB/s):
// bf16 operands everywhere (v_wmma_f32_16x16x32_bf16), fp32 accum/residual.
// Tile staging uses gfx1250 async global->LDS (ASYNCcnt) with K-loop
// double buffering; all weights pre-transposed to [N][K] bf16 so every
// stage is b128 asyncs only (3 per thread per GEMM stage).
// ---------------------------------------------------------------------------

typedef unsigned short u16;
typedef __attribute__((ext_vector_type(16))) __bf16 bf16x16;
typedef __attribute__((ext_vector_type(8)))  float  f32x8;

#define EMB  512
#define TSEQ 128
#define FFN  2048

__device__ __forceinline__ u16 f2bf(float f) {
  unsigned u = __float_as_uint(f);
  u += 0x7FFFu + ((u >> 16) & 1u);      // round-to-nearest-even
  return (u16)(u >> 16);
}

union Vec256 { uint4 q[2]; bf16x16 v; u16 u[16]; };

// A-operand (16x32 bf16, MxK): lane = hi*16+ln holds row M=ln,
// element e -> K = hi*16+e.
__device__ __forceinline__ bf16x16 make_a(const u16* rowp, int hi) {
  Vec256 t;
  t.q[0] = *(const uint4*)(rowp + hi * 16);
  t.q[1] = *(const uint4*)(rowp + hi * 16 + 8);
  return t.v;
}
// B-operand (32x16 bf16, KxN): lane holds col N=ln, element e -> K = hi*16+e.
// p points at 16 contiguous K values (B staged transposed [n][k] in LDS).
__device__ __forceinline__ bf16x16 make_b(const u16* p) {
  Vec256 t;
  t.q[0] = *(const uint4*)(p);
  t.q[1] = *(const uint4*)(p + 8);
  return t.v;
}

// ---- gfx1250 async global->LDS (no clang builtin; CDNA5 ISA) -----------
// Generic pointer to __shared__ has the LDS byte offset in its low 32 bits
// (flat-aperture rule, CDNA5 ISA).
__device__ __forceinline__ void async_b128(void* lds_dst, const void* gsrc) {
  unsigned lds = (unsigned)(unsigned long long)(uintptr_t)lds_dst;
  asm volatile("global_load_async_to_lds_b128 %0, %1, off"
               :: "v"(lds), "v"(gsrc) : "memory");
}
template <int N>
__device__ __forceinline__ void wait_async() {
  asm volatile("s_wait_asynccnt %0" :: "i"(N) : "memory");
}

// ---------------------------------------------------------------------------
// Tiled transpose + fp32->bf16: out[N][K] = (bf16) in[K][N]   (weights, once)
// ---------------------------------------------------------------------------
__global__ __launch_bounds__(256)
void k_transpose_bf16(const float* __restrict__ in, u16* __restrict__ out,
                      int K, int N) {
  __shared__ float tile[32][33];
  const int tx = threadIdx.x & 31, ty = threadIdx.x >> 5;   // 32 x 8
  const int n0 = blockIdx.x * 32, k0 = blockIdx.y * 32;
#pragma unroll
  for (int i = 0; i < 32; i += 8)
    tile[ty + i][tx] = in[(long)(k0 + ty + i) * N + n0 + tx];
  __syncthreads();
#pragma unroll
  for (int i = 0; i < 32; i += 8)
    out[(long)(n0 + ty + i) * K + k0 + tx] = f2bf(tile[tx][ty + i]);
}

// ---------------------------------------------------------------------------
// LayerNorm: one wave per token (E=512 -> 16 floats/lane), bf16 output.
// ---------------------------------------------------------------------------
struct u16x4 { u16 a, b, c, d; };

__global__ __launch_bounds__(256)
void k_layernorm_bf16(const float* __restrict__ x, const float* __restrict__ g,
                      const float* __restrict__ b, u16* __restrict__ h) {
  const int wave = threadIdx.x >> 5, lane = threadIdx.x & 31;
  const long token = (long)blockIdx.x * 8 + wave;
  const float* row = x + token * EMB;
  float4 vals[4];
  float s = 0.f, s2 = 0.f;
#pragma unroll
  for (int i = 0; i < 4; ++i) {
    vals[i] = *(const float4*)(row + i * 128 + lane * 4);
    s  += vals[i].x + vals[i].y + vals[i].z + vals[i].w;
    s2 += vals[i].x * vals[i].x + vals[i].y * vals[i].y +
          vals[i].z * vals[i].z + vals[i].w * vals[i].w;
  }
#pragma unroll
  for (int off = 16; off >= 1; off >>= 1) {
    s  += __shfl_xor(s,  off, 32);
    s2 += __shfl_xor(s2, off, 32);
  }
  const float mean = s * (1.0f / EMB);
  const float var  = s2 * (1.0f / EMB) - mean * mean;   // biased var (jnp.var)
  const float rstd = rsqrtf(var + 1e-5f);
  u16* hrow = h + token * EMB;
#pragma unroll
  for (int i = 0; i < 4; ++i) {
    const int c = i * 128 + lane * 4;
    u16x4 o;
    o.a = f2bf((vals[i].x - mean) * rstd * g[c + 0] + b[c + 0]);
    o.b = f2bf((vals[i].y - mean) * rstd * g[c + 1] + b[c + 1]);
    o.c = f2bf((vals[i].z - mean) * rstd * g[c + 2] + b[c + 2]);
    o.d = f2bf((vals[i].w - mean) * rstd * g[c + 3] + b[c + 3]);
    *(u16x4*)(hrow + c) = o;
  }
}

// ---------------------------------------------------------------------------
// GEMM: C[M,N] = A[M,K](bf16) @ W + bias, W given TRANSPOSED: BT[N][K] bf16.
// Block 256 = 8 waves, tile 128(M) x 64(N), wave tile 32x32 (2x2 WMMA).
// Async double-buffered K loop (3 asyncs/thread/stage).
// MODE 0: bf16 store. MODE 1: exact GELU -> bf16. MODE 2: fp32 += (residual).
// MODE 3: bf16 store transposed per 128-token batch: [b][col][tok&127].
// ---------------------------------------------------------------------------
template <int MODE>
__global__ __launch_bounds__(256)
void k_gemm_bf16(const u16* __restrict__ A, const u16* __restrict__ BT,
                 const float* __restrict__ bias, u16* __restrict__ outb,
                 float* __restrict__ outf, int N, int K) {
  __shared__ __attribute__((aligned(16))) u16 As[2][128][48];  // [m][k], 96B rows
  __shared__ __attribute__((aligned(16))) u16 Bs[2][64][48];   // [n][k]
  const int tid = threadIdx.x;
  const int wave = tid >> 5, lane = tid & 31, hi = lane >> 4, ln = lane & 15;
  const int wm = wave >> 1, wn = wave & 1;
  const long m0 = (long)blockIdx.y * 128;
  const int  n0 = blockIdx.x * 64;

  f32x8 acc[2][2];
#pragma unroll
  for (int ms = 0; ms < 2; ++ms)
#pragma unroll
    for (int ns = 0; ns < 2; ++ns) acc[ms][ns] = (f32x8)(0.0f);

  auto stage = [&](int kk, int buf) {
#pragma unroll
    for (int i = 0; i < 2; ++i) {                       // A tile 128x32
      const int j = tid * 2 + i;
      const int row = j >> 2, c4 = j & 3;
      async_b128(&As[buf][row][c4 * 8],
                 A + (m0 + row) * (long)K + kk * 32 + c4 * 8);
    }
    {                                                   // B tile 64x32 (from BT)
      const int n = tid >> 2, c4 = tid & 3;
      async_b128(&Bs[buf][n][c4 * 8],
                 BT + (long)(n0 + n) * K + kk * 32 + c4 * 8);
    }
  };

  const int ksteps = K >> 5;
  stage(0, 0);
  for (int kk = 0; kk < ksteps; ++kk) {
    const int buf = kk & 1;
    if (kk + 1 < ksteps) { stage(kk + 1, buf ^ 1); wait_async<3>(); }
    else                 { wait_async<0>(); }
    __syncthreads();

    bf16x16 av[2], bv[2];
#pragma unroll
    for (int ms = 0; ms < 2; ++ms)
      av[ms] = make_a(&As[buf][wm * 32 + ms * 16 + ln][0], hi);
#pragma unroll
    for (int ns = 0; ns < 2; ++ns)
      bv[ns] = make_b(&Bs[buf][wn * 32 + ns * 16 + ln][hi * 16]);
#pragma unroll
    for (int ms = 0; ms < 2; ++ms)
#pragma unroll
      for (int ns = 0; ns < 2; ++ns)
        acc[ms][ns] = __builtin_amdgcn_wmma_f32_16x16x32_bf16(
            false, av[ms], false, bv[ns], (short)0, acc[ms][ns], false, false);
    __syncthreads();   // protect buf before it is re-staged
  }

  // epilogue: D f32 layout -> row = base + r + hi*8, col = base + ln
#pragma unroll
  for (int ms = 0; ms < 2; ++ms)
#pragma unroll
    for (int ns = 0; ns < 2; ++ns)
#pragma unroll
      for (int r = 0; r < 8; ++r) {
        const long row = m0 + wm * 32 + ms * 16 + r + hi * 8;
        const int  col = n0 + wn * 32 + ns * 16 + ln;
        float val = acc[ms][ns][r] + bias[col];
        if (MODE == 0) {
          outb[row * N + col] = f2bf(val);
        } else if (MODE == 1) {
          val = 0.5f * val * (1.0f + erff(val * 0.70710678118f));  // exact GELU
          outb[row * N + col] = f2bf(val);
        } else if (MODE == 2) {
          outf[row * N + col] += val;      // fused residual into d_out
        } else {                           // MODE 3: per-batch transposed store
          const long batch = row >> 7;
          const int  key   = (int)(row & 127);
          outb[batch * (128L * EMB) + (long)col * 128 + key] = f2bf(val);
        }
      }
}

// ---------------------------------------------------------------------------
// Causal attention, one workgroup per batch (T=128, E=512).
// S in registers (8 v8f / wave); softmax via half-wave shfl_xor;
// P -> LDS bf16; P@V from pre-transposed V^T with async staging.
// out = x + attn (fused residual).
// ---------------------------------------------------------------------------
__global__ __launch_bounds__(256)
void k_attention(const u16* __restrict__ q, const u16* __restrict__ k,
                 const u16* __restrict__ vT, const float* __restrict__ x,
                 float* __restrict__ out) {
  __shared__ __attribute__((aligned(16))) union {
    struct { u16 qt[128][48]; u16 kt[128][48]; } p1;   // 24 KB
    struct { u16 P[128][136]; u16 vt[128][48]; } p2;   // 46 KB
  } sm;
  const int tid = threadIdx.x, wave = tid >> 5, lane = tid & 31;
  const int hi = lane >> 4, ln = lane & 15;
  const long base = (long)blockIdx.x * TSEQ * EMB;
  const u16* qb  = q  + base;
  const u16* kb  = k  + base;
  const u16* vTb = vT + base;   // [emb][key] per batch

  // ---- Phase 1: S = Q @ K^T, rows 16*wave..16*wave+15 per wave ----
  f32x8 s[8];
#pragma unroll
  for (int t = 0; t < 8; ++t) s[t] = (f32x8)(0.0f);

  for (int kk = 0; kk < EMB / 32; ++kk) {
    __syncthreads();
#pragma unroll
    for (int i = 0; i < 2; ++i) {
      const int j = tid * 2 + i;
      const int row = j >> 2, c4 = j & 3;
      async_b128(&sm.p1.qt[row][c4 * 8], qb + row * EMB + kk * 32 + c4 * 8);
      async_b128(&sm.p1.kt[row][c4 * 8], kb + row * EMB + kk * 32 + c4 * 8);
    }
    wait_async<0>();
    __syncthreads();
    const bf16x16 av = make_a(&sm.p1.qt[wave * 16 + ln][0], hi);
#pragma unroll
    for (int t = 0; t < 8; ++t) {
      const bf16x16 bv = make_b(&sm.p1.kt[t * 16 + ln][hi * 16]);
      s[t] = __builtin_amdgcn_wmma_f32_16x16x32_bf16(
          false, av, false, bv, (short)0, s[t], false, false);
    }
  }

  // ---- Phase 2: scale, causal mask, softmax entirely in registers ----
  const float scale = 0.125f;   // 1/sqrt(EMB_S=64)
  float rsum[8];
#pragma unroll
  for (int r = 0; r < 8; ++r) {
    const int mrow = wave * 16 + r + hi * 8;     // query index
    float mx = -1e30f;
#pragma unroll
    for (int t = 0; t < 8; ++t) {
      const int col = t * 16 + ln;               // key index
      float val = s[t][r] * scale;
      if (col > mrow) val = -1e30f;              // causal mask
      s[t][r] = val;
      mx = fmaxf(mx, val);
    }
    // row lives in one 16-lane half: xor masks 1,2,4,8 stay in the half
#pragma unroll
    for (int off = 1; off <= 8; off <<= 1) mx = fmaxf(mx, __shfl_xor(mx, off, 32));
    float su = 0.f;
#pragma unroll
    for (int t = 0; t < 8; ++t) {
      const float p = __expf(s[t][r] - mx);
      s[t][r] = p;
      su += p;
    }
#pragma unroll
    for (int off = 1; off <= 8; off <<= 1) su += __shfl_xor(su, off, 32);
    rsum[r] = su;
  }
  __syncthreads();              // done reading qt/kt (union reuse)
#pragma unroll
  for (int r = 0; r < 8; ++r) {
    const float inv = 1.0f / rsum[r];
    const int mrow = wave * 16 + r + hi * 8;
#pragma unroll
    for (int t = 0; t < 8; ++t)
      sm.p2.P[mrow][t * 16 + ln] = f2bf(s[t][r] * inv);
  }
  __syncthreads();

  // ---- Phase 3: Y = P @ V (V^T pre-transposed), fused residual ----
  for (int nc = 0; nc < 4; ++nc) {              // 4 chunks of 128 emb cols
    f32x8 y[8];
#pragma unroll
    for (int t = 0; t < 8; ++t) y[t] = (f32x8)(0.0f);
    for (int kt = 0; kt < 4; ++kt) {            // 4 chunks of 32 keys
      __syncthreads();
#pragma unroll
      for (int i = 0; i < 2; ++i) {             // vt[n][k] <- vT[nc*128+n][kt*32+k]
        const int j = tid * 2 + i;
        const int n = j >> 2, c4 = j & 3;
        async_b128(&sm.p2.vt[n][c4 * 8],
                   vTb + (long)(nc * 128 + n) * 128 + kt * 32 + c4 * 8);
      }
      wait_async<0>();
      __syncthreads();
      const bf16x16 av = make_a(&sm.p2.P[wave * 16 + ln][kt * 32], hi);
#pragma unroll
      for (int t = 0; t < 8; ++t) {
        const bf16x16 bv = make_b(&sm.p2.vt[t * 16 + ln][hi * 16]);
        y[t] = __builtin_amdgcn_wmma_f32_16x16x32_bf16(
            false, av, false, bv, (short)0, y[t], false, false);
      }
    }
#pragma unroll
    for (int t = 0; t < 8; ++t)
#pragma unroll
      for (int r = 0; r < 8; ++r) {
        const long idx = base + (long)(wave * 16 + r + hi * 8) * EMB +
                         nc * 128 + t * 16 + ln;
        out[idx] = x[idx] + y[t][r];
      }
  }
}

// ---------------------------------------------------------------------------
// Host-side orchestration
// ---------------------------------------------------------------------------
extern "C" void kernel_launch(void* const* d_in, const int* in_sizes, int n_in,
                              void* d_out, int out_size, void* d_ws, size_t ws_size,
                              hipStream_t stream) {
  const float* x   = (const float*)d_in[0];
  const float* wq  = (const float*)d_in[1];
  const float* bq  = (const float*)d_in[2];
  const float* wk  = (const float*)d_in[3];
  const float* bk  = (const float*)d_in[4];
  const float* wv  = (const float*)d_in[5];
  const float* bv  = (const float*)d_in[6];
  const float* g1  = (const float*)d_in[7];
  const float* b1  = (const float*)d_in[8];
  const float* g2  = (const float*)d_in[9];
  const float* b2  = (const float*)d_in[10];
  const float* w1  = (const float*)d_in[11];
  const float* bm1 = (const float*)d_in[12];
  const float* w2  = (const float*)d_in[13];
  const float* bm2 = (const float*)d_in[14];
  float* out = (float*)d_out;

  const long NTOK = 512L * TSEQ;   // 65536 tokens

  char* ws = (char*)d_ws;
  size_t off = 0;
  auto alloc = [&](size_t bytes) -> void* {
    void* p = ws + off;
    off += (bytes + 255) & ~(size_t)255;
    return p;
  };
  u16* wqT = (u16*)alloc((size_t)EMB * EMB * 2);   // [N][K] bf16
  u16* wkT = (u16*)alloc((size_t)EMB * EMB * 2);
  u16* wvT = (u16*)alloc((size_t)EMB * EMB * 2);
  u16* w1T = (u16*)alloc((size_t)FFN * EMB * 2);   // [FFN][EMB]
  u16* w2T = (u16*)alloc((size_t)EMB * FFN * 2);   // [EMB][FFN]
  u16* h_b = (u16*)alloc((size_t)NTOK * EMB * 2);
  u16* q_b = (u16*)alloc((size_t)NTOK * EMB * 2);
  u16* k_b = (u16*)alloc((size_t)NTOK * EMB * 2);
  u16* vT_b = (u16*)alloc((size_t)NTOK * EMB * 2); // per-batch [emb][key]
  u16* m1_b = (u16*)alloc((size_t)NTOK * FFN * 2);
  (void)ws_size; (void)n_in; (void)in_sizes; (void)out_size;

  // 1) weights: fp32 [K][N] -> bf16 transposed [N][K]
  k_transpose_bf16<<<dim3(EMB / 32, EMB / 32), 256, 0, stream>>>(wq, wqT, EMB, EMB);
  k_transpose_bf16<<<dim3(EMB / 32, EMB / 32), 256, 0, stream>>>(wk, wkT, EMB, EMB);
  k_transpose_bf16<<<dim3(EMB / 32, EMB / 32), 256, 0, stream>>>(wv, wvT, EMB, EMB);
  k_transpose_bf16<<<dim3(FFN / 32, EMB / 32), 256, 0, stream>>>(w1, w1T, EMB, FFN);
  k_transpose_bf16<<<dim3(EMB / 32, FFN / 32), 256, 0, stream>>>(w2, w2T, FFN, EMB);

  // 2) h = LN1(x)  (bf16)
  k_layernorm_bf16<<<NTOK / 8, 256, 0, stream>>>(x, g1, b1, h_b);

  // 3) q,k = h @ W + b (bf16); v stored transposed per batch (MODE 3)
  {
    dim3 grid(EMB / 64, NTOK / 128);
    k_gemm_bf16<0><<<grid, 256, 0, stream>>>(h_b, wqT, bq, q_b, nullptr, EMB, EMB);
    k_gemm_bf16<0><<<grid, 256, 0, stream>>>(h_b, wkT, bk, k_b, nullptr, EMB, EMB);
    k_gemm_bf16<3><<<grid, 256, 0, stream>>>(h_b, wvT, bv, vT_b, nullptr, EMB, EMB);
  }

  // 4) d_out = x + causal_attention(q,k,v)
  k_attention<<<512, 256, 0, stream>>>(q_b, k_b, vT_b, x, out);

  // 5) h = LN2(d_out)  (bf16, reuse h buffer)
  k_layernorm_bf16<<<NTOK / 8, 256, 0, stream>>>(out, g2, b2, h_b);

  // 6) m1 = gelu(h @ w1 + bm1)  (bf16)
  k_gemm_bf16<1><<<dim3(FFN / 64, NTOK / 128), 256, 0, stream>>>(
      h_b, w1T, bm1, m1_b, nullptr, FFN, EMB);

  // 7) d_out += m1 @ w2 + bm2
  k_gemm_bf16<2><<<dim3(EMB / 64, NTOK / 128), 256, 0, stream>>>(
      m1_b, w2T, bm2, nullptr, out, EMB, FFN);
}
